// arrow_60816736911489
// MI455X (gfx1250) — matrix operand order repaired and checked by reading, not compiled, not run on hardware
//
#include <hip/hip_runtime.h>
#include <hip/hip_bf16.h>

// ---------------------------------------------------------------------------
// CDNA5 (gfx1250) DiT block.
//  - All GEMMs + flash attention on v_wmma_f32_16x16x32_bf16 (f32 accum)
//  - LDS staging via GLOBAL_LOAD_ASYNC_TO_LDS_B128 (ASYNCcnt) when available,
//    double-buffered to overlap HBM/L2 latency with the matrix pipe.
//  - Weights pre-transposed to [N][K] bf16 and V stored [B,H,D,L] so every
//    LDS stage is a contiguous b128 copy (no in-kernel transposes).
// ---------------------------------------------------------------------------

typedef __attribute__((ext_vector_type(16))) __bf16 v16bf;
typedef __attribute__((ext_vector_type(8)))  float  v8f;
typedef __attribute__((ext_vector_type(4)))  int    v4i_t;

#define WMMA_BF16(A, B, C) \
  __builtin_amdgcn_wmma_f32_16x16x32_bf16(false, (A), false, (B), (short)0, (C), false, false)

#if defined(__AMDGCN__) && __has_builtin(__builtin_amdgcn_global_load_async_to_lds_b128) && \
    __has_builtin(__builtin_amdgcn_s_wait_asynccnt)
#define USE_ASYNC_LDS 1
#else
#define USE_ASYNC_LDS 0
#endif

#define AS1 __attribute__((address_space(1)))
#define AS3 __attribute__((address_space(3)))

// per-lane 16-byte global -> LDS copy
__device__ __forceinline__ void async_cp128(unsigned short* ldst,
                                            const unsigned short* gsrc) {
#if USE_ASYNC_LDS
  // builtin signature (from hipcc diagnostic): (v4i AS1*, v4i AS3*, Ii offset, Ii cpol)
  unsigned short* g = (unsigned short*)gsrc;   // strip const
  __builtin_amdgcn_global_load_async_to_lds_b128((AS1 v4i_t*)g, (AS3 v4i_t*)ldst, 0, 0);
#else
  *(uint4*)ldst = *(const uint4*)gsrc;
#endif
}
__device__ __forceinline__ void wait_async_all() {
#if USE_ASYNC_LDS
  __builtin_amdgcn_s_wait_asynccnt(0);
#endif
}

__device__ __forceinline__ unsigned short f2bf(float f) {
  unsigned int u = __float_as_uint(f);
  u += 0x7FFFu + ((u >> 16) & 1u);        // round-to-nearest-even
  return (unsigned short)(u >> 16);
}
__device__ __forceinline__ float bf2f(unsigned short h) {
  return __uint_as_float(((unsigned int)h) << 16);
}

union AFrag { v16bf v; uint4 q[2]; };      // 16 bf16 = one WMMA A/B fragment

__device__ __forceinline__ v8f vzero8() {
  v8f z;
#pragma unroll
  for (int i = 0; i < 8; ++i) z[i] = 0.f;
  return z;
}

// ------------------------------ elementwise --------------------------------

// f32 [R][C] -> bf16 [C][R]  (one-time weight transpose+cast)
__global__ void k_cast_bf16_t(const float* __restrict__ src,
                              unsigned short* __restrict__ dst, int R, int C) {
  int i = blockIdx.x * blockDim.x + threadIdx.x;
  int r = i / C, c = i - r * C;
  dst[(size_t)c * R + r] = f2bf(src[i]);
}

// mod = silu(c) @ ada_w + ada_b   (2 x 1024 x 6144; tiny, VALU is fine)
__global__ void k_ada(const float* __restrict__ c, const float* __restrict__ w,
                      const float* __restrict__ b, float* __restrict__ mod) {
  __shared__ float sc[1024];
  int bb = blockIdx.y;
  for (int k = threadIdx.x; k < 1024; k += blockDim.x) {
    float v = c[bb * 1024 + k];
    sc[k] = v / (1.f + expf(-v));
  }
  __syncthreads();
  int j = blockIdx.x * blockDim.x + threadIdx.x;
  float acc = b[j];
  for (int k = 0; k < 1024; ++k) acc += sc[k] * w[k * 6144 + j];
  mod[bb * 6144 + j] = acc;
}

// h = rms_norm(x, w) * (1 + scale) + shift  -> bf16   (one block per row)
__global__ void k_rmsmod(const float* __restrict__ x, const float* __restrict__ w,
                         const float* __restrict__ mod, int shiftOfs, int scaleOfs,
                         unsigned short* __restrict__ out) {
  int row = blockIdx.x;            // 0..4095 == b*2048 + l
  int bb  = row >> 11;
  const float* xr = x + (size_t)row * 1024;
  float xv[4];
  float ss = 0.f;
#pragma unroll
  for (int i = 0; i < 4; ++i) { xv[i] = xr[threadIdx.x + 256 * i]; ss += xv[i] * xv[i]; }
  for (int m = 1; m < 32; m <<= 1) ss += __shfl_xor(ss, m, 32);
  __shared__ float red[8];
  int wid = threadIdx.x >> 5, lane = threadIdx.x & 31;
  if (lane == 0) red[wid] = ss;
  __syncthreads();
  float tot = red[0];
#pragma unroll
  for (int i = 1; i < 8; ++i) tot += red[i];
  float rn = rsqrtf(tot * (1.f / 1024.f) + 1e-6f);
  const float* sh  = mod + bb * 6144 + shiftOfs;
  const float* sca = mod + bb * 6144 + scaleOfs;
#pragma unroll
  for (int i = 0; i < 4; ++i) {
    int col = threadIdx.x + 256 * i;
    float h = xv[i] * rn * w[col] * (1.f + sca[col]) + sh[col];
    out[(size_t)row * 1024 + col] = f2bf(h);
  }
}

// out = x + gate * y
__global__ void k_residual(const float* __restrict__ x, const float* __restrict__ y,
                           const float* __restrict__ mod, int gateOfs,
                           float* __restrict__ out) {
  int i   = blockIdx.x * blockDim.x + threadIdx.x;
  int col = i & 1023;
  int b   = i >> 21;               // / (2048*1024)
  out[i] = x[i] + mod[b * 6144 + gateOfs + col] * y[i];
}

// RoPE + rearrange: qkv [B,L,3,H,D] bf16 -> q,k [B,H,L,D], v [B,H,D,L]
__global__ void k_rope(const unsigned short* __restrict__ qkv,
                       const float* __restrict__ fcos, const float* __restrict__ fsin,
                       unsigned short* __restrict__ q, unsigned short* __restrict__ k,
                       unsigned short* __restrict__ vt) {
  int idx = blockIdx.x * blockDim.x + threadIdx.x;   // B*L*H*32 threads
  int i = idx & 31;
  int h = (idx >> 5) & 15;
  int l = (idx >> 9) & 2047;
  int b = idx >> 20;
  float cc = fcos[l * 32 + i], ss = fsin[l * 32 + i];
  size_t srow = ((size_t)(b * 2048 + l)) * 3072;
  size_t bh   = (size_t)(b * 16 + h);
  size_t dst  = (bh * 2048 + l) * 64 + 2 * i;
  {
    size_t s0 = srow + 0 * 1024 + h * 64 + 2 * i;
    float t1 = bf2f(qkv[s0]), t2 = bf2f(qkv[s0 + 1]);
    q[dst] = f2bf(t1 * cc - t2 * ss);  q[dst + 1] = f2bf(t1 * ss + t2 * cc);
  }
  {
    size_t s0 = srow + 1024 + h * 64 + 2 * i;
    float t1 = bf2f(qkv[s0]), t2 = bf2f(qkv[s0 + 1]);
    k[dst] = f2bf(t1 * cc - t2 * ss);  k[dst + 1] = f2bf(t1 * ss + t2 * cc);
  }
  {
    size_t s0 = srow + 2048 + h * 64 + 2 * i;
    vt[(bh * 64 + 2 * i) * 2048 + l]     = qkv[s0];      // d-major for attention
    vt[(bh * 64 + 2 * i + 1) * 2048 + l] = qkv[s0 + 1];
  }
}

// ------------------------------ WMMA GEMM ----------------------------------
// C[M,N] = A[M,K](bf16,row) @ Wt[N,K](bf16, pre-transposed) + bias
// mode 0: f32 out   mode 1: gelu(tanh) -> bf16   mode 2: bf16 out
// Block 256 thr (8 waves), tile 128x128, BK=32; waves as 4(M)x2(N), 32x64 each.
// Double-buffered LDS; stages are contiguous async b128 copies.
__global__ __launch_bounds__(256)
void k_gemm_bf16(const unsigned short* __restrict__ A,
                 const unsigned short* __restrict__ Wt,
                 const float* __restrict__ bias,
                 float* __restrict__ outF, unsigned short* __restrict__ outH,
                 int M, int N, int K, int mode) {
  __shared__ unsigned short sA[2][128 * 32];   // [m][k]
  __shared__ unsigned short sB[2][128 * 32];   // [n][k]
  const int tid  = threadIdx.x;
  const int lane = tid & 31;
  const int w    = tid >> 5;
  const int m0 = blockIdx.y * 128, n0 = blockIdx.x * 128;
  const int mb = (w >> 1) * 32,    nb = (w & 1) * 64;
  const int lo = lane & 15,        hi = lane >> 4;

  const int srow = tid >> 1, skc = (tid & 1) * 16;   // 128 rows x 32, 16/thr
  const unsigned short* aSrc = A  + (size_t)(m0 + srow) * K + skc;
  const unsigned short* bSrc = Wt + (size_t)(n0 + srow) * K + skc;

  auto stage = [&](int buf, int k0) {
    async_cp128(&sA[buf][srow * 32 + skc],     aSrc + k0);
    async_cp128(&sA[buf][srow * 32 + skc + 8], aSrc + k0 + 8);
    async_cp128(&sB[buf][srow * 32 + skc],     bSrc + k0);
    async_cp128(&sB[buf][srow * 32 + skc + 8], bSrc + k0 + 8);
  };

  v8f acc[2][4];
#pragma unroll
  for (int i = 0; i < 2; ++i)
#pragma unroll
    for (int j = 0; j < 4; ++j) acc[i][j] = vzero8();

  stage(0, 0);
  wait_async_all();
  __syncthreads();

  int cur = 0;
  for (int k0 = 0; k0 < K; k0 += 32) {
    if (k0 + 32 < K) stage(cur ^ 1, k0 + 32);   // prefetch next tile (async)

    AFrag af[2];
#pragma unroll
    for (int ms = 0; ms < 2; ++ms) {            // A frag: lanes 0-15 K0-7/16-23
      int r = mb + ms * 16 + lo;                //         lanes16-31 K8-15/24-31
      af[ms].q[0] = *(const uint4*)&sA[cur][r * 32 + hi * 8];
      af[ms].q[1] = *(const uint4*)&sA[cur][r * 32 + 16 + hi * 8];
    }
#pragma unroll
    for (int ns = 0; ns < 4; ++ns) {
      AFrag bfr;                                // B frag: lane = column
      int c = nb + ns * 16 + lo;
      bfr.q[0] = *(const uint4*)&sB[cur][c * 32 + hi * 8];
      bfr.q[1] = *(const uint4*)&sB[cur][c * 32 + 16 + hi * 8];
#pragma unroll
      for (int ms = 0; ms < 2; ++ms)
        acc[ms][ns] = WMMA_BF16(af[ms].v, bfr.v, acc[ms][ns]);
    }

    wait_async_all();
    __syncthreads();
    cur ^= 1;
  }

  // epilogue (mode branch hoisted out of unrolled stores)
  if (mode == 0) {
#pragma unroll
    for (int ms = 0; ms < 2; ++ms)
#pragma unroll
      for (int ns = 0; ns < 4; ++ns) {
        int colg = n0 + nb + ns * 16 + lo;
        float bv = bias[colg];
#pragma unroll
        for (int r = 0; r < 8; ++r) {
          int rowg = m0 + mb + ms * 16 + r + 8 * hi;
          outF[(size_t)rowg * N + colg] = acc[ms][ns][r] + bv;
        }
      }
  } else if (mode == 1) {
#pragma unroll
    for (int ms = 0; ms < 2; ++ms)
#pragma unroll
      for (int ns = 0; ns < 4; ++ns) {
        int colg = n0 + nb + ns * 16 + lo;
        float bv = bias[colg];
#pragma unroll
        for (int r = 0; r < 8; ++r) {
          int rowg = m0 + mb + ms * 16 + r + 8 * hi;
          float v = acc[ms][ns][r] + bv;
          float g = 0.5f * v * (1.f + tanhf(0.7978845608028654f * (v + 0.044715f * v * v * v)));
          outH[(size_t)rowg * N + colg] = f2bf(g);
        }
      }
  } else {
#pragma unroll
    for (int ms = 0; ms < 2; ++ms)
#pragma unroll
      for (int ns = 0; ns < 4; ++ns) {
        int colg = n0 + nb + ns * 16 + lo;
        float bv = bias[colg];
#pragma unroll
        for (int r = 0; r < 8; ++r) {
          int rowg = m0 + mb + ms * 16 + r + 8 * hi;
          outH[(size_t)rowg * N + colg] = f2bf(acc[ms][ns][r] + bv);
        }
      }
  }
}

// --------------------------- WMMA flash attention --------------------------
// One block per (64 query rows, b*16+h). 4 waves x 16 rows. Online softmax.
// K [B,H,L,D], V pre-transposed [B,H,D,L]; double-buffered async staging.
__global__ __launch_bounds__(128)
void k_attn(const unsigned short* __restrict__ q,
            const unsigned short* __restrict__ k,
            const unsigned short* __restrict__ vt,
            unsigned short* __restrict__ out) {
  __shared__ unsigned short sK[2][32 * 64];      // [key][d]
  __shared__ unsigned short sV[2][64 * 32];      // [d][key]
  __shared__ unsigned short sP[4 * 16 * 32];     // per-wave P (C->A layout fix)
  const int tid  = threadIdx.x;
  const int lane = tid & 31;
  const int w    = tid >> 5;
  const int lo = lane & 15, hi = lane >> 4;
  const int bh = blockIdx.y;                     // b*16 + h
  const int b  = bh >> 4, h = bh & 15;
  const int q0 = blockIdx.x * 64 + w * 16;
  const size_t head = ((size_t)bh) * 2048 * 64;

  const unsigned short* kp = k  + head;
  const unsigned short* vp = vt + head;

  const int krow = tid >> 2, kch = (tid & 3) * 16;   // K: 32 rows x 64
  const int vrow = tid >> 1, vch = (tid & 1) * 16;   // V: 64 rows x 32
  auto stage = [&](int buf, int j) {
    const unsigned short* ks = kp + (size_t)(j + krow) * 64 + kch;
    async_cp128(&sK[buf][krow * 64 + kch],     ks);
    async_cp128(&sK[buf][krow * 64 + kch + 8], ks + 8);
    const unsigned short* vs = vp + (size_t)vrow * 2048 + j + vch;
    async_cp128(&sV[buf][vrow * 32 + vch],     vs);
    async_cp128(&sV[buf][vrow * 32 + vch + 8], vs + 8);
  };

  AFrag qa[2];
  {
    const unsigned short* qp = q + head + (size_t)(q0 + lo) * 64;
#pragma unroll
    for (int kt = 0; kt < 2; ++kt) {
      qa[kt].q[0] = *(const uint4*)(qp + kt * 32 + hi * 8);
      qa[kt].q[1] = *(const uint4*)(qp + kt * 32 + 16 + hi * 8);
    }
  }

  v8f o[4];
#pragma unroll
  for (int i = 0; i < 4; ++i) o[i] = vzero8();
  float mrow[8], lrow[8];
#pragma unroll
  for (int r = 0; r < 8; ++r) { mrow[r] = -3.0e38f; lrow[r] = 0.f; }

  const int wp = w * 16 * 32;

  stage(0, 0);
  wait_async_all();
  __syncthreads();

  int cur = 0;
  for (int j = 0; j < 2048; j += 32) {
    if (j + 32 < 2048) stage(cur ^ 1, j + 32);   // prefetch next key block

    // S(16x32) = Q(16x64) @ K^T, two 16-key tiles, two k=32 WMMAs each
    v8f s0 = vzero8(), s1 = vzero8();
#pragma unroll
    for (int nt = 0; nt < 2; ++nt) {
      v8f st = vzero8();
#pragma unroll
      for (int kt = 0; kt < 2; ++kt) {
        AFrag kb;
        int c = nt * 16 + lo;
        kb.q[0] = *(const uint4*)&sK[cur][c * 64 + kt * 32 + hi * 8];
        kb.q[1] = *(const uint4*)&sK[cur][c * 64 + kt * 32 + 16 + hi * 8];
        st = WMMA_BF16(qa[kt].v, kb.v, st);
      }
      if (nt == 0) s0 = st; else s1 = st;
    }

    // online softmax (row = r + 8*hi lives across 16 lanes of one VGPR)
#pragma unroll
    for (int r = 0; r < 8; ++r) {
      float a0 = s0[r] * 0.125f, a1 = s1[r] * 0.125f;   // 1/sqrt(64)
      float mx = fmaxf(a0, a1);
      for (int mk = 1; mk < 16; mk <<= 1) mx = fmaxf(mx, __shfl_xor(mx, mk, 16));
      float mn    = fmaxf(mrow[r], mx);
      float alpha = expf(mrow[r] - mn);
      float p0 = expf(a0 - mn), p1 = expf(a1 - mn);
      float ps = p0 + p1;
      for (int mk = 1; mk < 16; mk <<= 1) ps += __shfl_xor(ps, mk, 16);
      lrow[r] = lrow[r] * alpha + ps;
      mrow[r] = mn;
#pragma unroll
      for (int t = 0; t < 4; ++t) o[t][r] *= alpha;
      int row = r + 8 * hi;
      sP[wp + row * 32 + lo]      = f2bf(p0);
      sP[wp + row * 32 + 16 + lo] = f2bf(p1);
    }

    // O(16x64) += P(16x32) @ V(32x64)
    AFrag pa;
    pa.q[0] = *(const uint4*)&sP[wp + lo * 32 + hi * 8];
    pa.q[1] = *(const uint4*)&sP[wp + lo * 32 + 16 + hi * 8];
#pragma unroll
    for (int ds = 0; ds < 4; ++ds) {
      AFrag vb;
      int c = ds * 16 + lo;
      vb.q[0] = *(const uint4*)&sV[cur][c * 32 + hi * 8];
      vb.q[1] = *(const uint4*)&sV[cur][c * 32 + 16 + hi * 8];
      o[ds] = WMMA_BF16(pa.v, vb.v, o[ds]);
    }

    wait_async_all();
    __syncthreads();
    cur ^= 1;
  }

  // epilogue: out[b, l, h*64+d] (bf16), normalized by row sum
#pragma unroll
  for (int ds = 0; ds < 4; ++ds)
#pragma unroll
    for (int r = 0; r < 8; ++r) {
      int l   = q0 + r + 8 * hi;
      int col = h * 64 + ds * 16 + lo;
      out[((size_t)(b * 2048 + l)) * 1024 + col] = f2bf(o[ds][r] / lrow[r]);
    }
}

// ------------------------------- launcher ----------------------------------

extern "C" void kernel_launch(void* const* d_in, const int* in_sizes, int n_in,
                              void* d_out, int out_size, void* d_ws, size_t ws_size,
                              hipStream_t stream) {
  (void)in_sizes; (void)n_in; (void)out_size; (void)ws_size;
  const float* x      = (const float*)d_in[0];
  const float* c      = (const float*)d_in[1];
  const float* fcos   = (const float*)d_in[2];
  const float* fsin   = (const float*)d_in[3];
  const float* rms1w  = (const float*)d_in[4];
  const float* rms2w  = (const float*)d_in[5];
  const float* qkv_w  = (const float*)d_in[6];
  const float* qkv_b  = (const float*)d_in[7];
  const float* proj_w = (const float*)d_in[8];
  const float* proj_b = (const float*)d_in[9];
  const float* mlp_w1 = (const float*)d_in[10];
  const float* mlp_b1 = (const float*)d_in[11];
  const float* mlp_w2 = (const float*)d_in[12];
  const float* mlp_b2 = (const float*)d_in[13];
  const float* ada_w  = (const float*)d_in[14];
  const float* ada_b  = (const float*)d_in[15];
  float* outp = (float*)d_out;

  char* ws = (char*)d_ws;
  size_t off = 0;
  auto alloc = [&](size_t bytes) {
    size_t r = off; off += (bytes + 255) & ~(size_t)255; return r;
  };

  // all weights stored transposed: Wt[N][K] bf16
  unsigned short* wQKV  = (unsigned short*)(ws + alloc((size_t)1024 * 3072 * 2));
  unsigned short* wPROJ = (unsigned short*)(ws + alloc((size_t)1024 * 1024 * 2));
  unsigned short* wM1   = (unsigned short*)(ws + alloc((size_t)1024 * 4096 * 2));
  unsigned short* wM2   = (unsigned short*)(ws + alloc((size_t)4096 * 1024 * 2));
  float*          modb  = (float*)         (ws + alloc((size_t)2 * 6144 * 4));
  unsigned short* hbuf  = (unsigned short*)(ws + alloc((size_t)4096 * 1024 * 2));
  unsigned short* qkvb  = (unsigned short*)(ws + alloc((size_t)4096 * 3072 * 2));
  unsigned short* qb    = (unsigned short*)(ws + alloc((size_t)4096 * 1024 * 2));
  unsigned short* kb    = (unsigned short*)(ws + alloc((size_t)4096 * 1024 * 2));
  unsigned short* vbt   = (unsigned short*)(ws + alloc((size_t)4096 * 1024 * 2));
  unsigned short* attno = (unsigned short*)(ws + alloc((size_t)4096 * 1024 * 2));
  float*          projo = (float*)         (ws + alloc((size_t)4096 * 1024 * 4));
  float*          x1    = (float*)         (ws + alloc((size_t)4096 * 1024 * 4));
  // Aliases (lifetimes disjoint): y1 reuses qkv+q region (33.5MB), y2 reuses k+v (16.8MB)
  unsigned short* y1 = qkvb;
  float*          y2 = (float*)kb;

  dim3 b256(256);
  // 1) weights -> bf16, transposed to [N][K]
  k_cast_bf16_t<<<(1024 * 3072) / 256, b256, 0, stream>>>(qkv_w, wQKV, 1024, 3072);
  k_cast_bf16_t<<<(1024 * 1024) / 256, b256, 0, stream>>>(proj_w, wPROJ, 1024, 1024);
  k_cast_bf16_t<<<(1024 * 4096) / 256, b256, 0, stream>>>(mlp_w1, wM1, 1024, 4096);
  k_cast_bf16_t<<<(4096 * 1024) / 256, b256, 0, stream>>>(mlp_w2, wM2, 4096, 1024);
  // 2) adaLN modulation
  k_ada<<<dim3(24, 2), b256, 0, stream>>>(c, ada_w, ada_b, modb);
  // 3) h = modulate(rms_norm(x)) -> bf16
  k_rmsmod<<<4096, b256, 0, stream>>>(x, rms1w, modb, 0, 1024, hbuf);
  // 4) qkv = h @ qkv_w + b  (bf16 out)
  k_gemm_bf16<<<dim3(3072 / 128, 4096 / 128), b256, 0, stream>>>(
      hbuf, wQKV, qkv_b, nullptr, qkvb, 4096, 3072, 1024, 2);
  // 5) RoPE + head-major rearrange (V stored d-major)
  k_rope<<<(2 * 2048 * 16 * 32) / 256, b256, 0, stream>>>(qkvb, fcos, fsin, qb, kb, vbt);
  // 6) flash attention (WMMA)
  k_attn<<<dim3(2048 / 64, 32), dim3(128), 0, stream>>>(qb, kb, vbt, attno);
  // 7) proj GEMM (f32 out)
  k_gemm_bf16<<<dim3(1024 / 128, 4096 / 128), b256, 0, stream>>>(
      attno, wPROJ, proj_b, projo, nullptr, 4096, 1024, 1024, 0);
  // 8) x1 = x + gate_msa * proj
  k_residual<<<(4096 * 1024) / 256, b256, 0, stream>>>(x, projo, modb, 2 * 1024, x1);
  // 9) h2 = modulate(rms_norm(x1))
  k_rmsmod<<<4096, b256, 0, stream>>>(x1, rms2w, modb, 3 * 1024, 4 * 1024, hbuf);
  // 10) mlp1 + gelu (bf16 out)
  k_gemm_bf16<<<dim3(4096 / 128, 4096 / 128), b256, 0, stream>>>(
      hbuf, wM1, mlp_b1, nullptr, y1, 4096, 4096, 1024, 1);
  // 11) mlp2 (f32 out)
  k_gemm_bf16<<<dim3(1024 / 128, 4096 / 128), b256, 0, stream>>>(
      y1, wM2, mlp_b2, y2, nullptr, 4096, 1024, 4096, 0);
  // 12) out = x1 + gate_mlp * y
  k_residual<<<(4096 * 1024) / 256, b256, 0, stream>>>(x1, y2, modb, 5 * 1024, outp);
}
